// Encoder_63307817943738
// MI455X (gfx1250) — matrix operand order repaired and checked by reading, not compiled, not run on hardware
//
#include <hip/hip_runtime.h>
#include <hip/hip_bf16.h>

// ---- CDNA5 vector types for WMMA ----
typedef _Float16 h16;
typedef h16   v16h __attribute__((ext_vector_type(16)));
typedef h16   v8h  __attribute__((ext_vector_type(8)));
typedef float v8f  __attribute__((ext_vector_type(8)));

#define BATCH  4096
#define TSTEPS 512
#define IN_DIM 22
#define HDIM   64
#define NG     256   // 4*H
#define MTILE  16

#define LOG2E 1.4426950408889634f

// Branchless fast activations: single v_exp_f32 + v_rcp_f32, no libm range splits.
__device__ __forceinline__ float fast_sigmoid(float x) {
    float e = __builtin_amdgcn_exp2f(-LOG2E * x);
    return __builtin_amdgcn_rcpf(1.0f + e);
}
__device__ __forceinline__ float fast_tanh(float x) {
    float e = __builtin_amdgcn_exp2f(-2.0f * LOG2E * x);
    return 2.0f * __builtin_amdgcn_rcpf(1.0f + e) - 1.0f;
}

// Build a 16x32 f16 A-fragment in this kernel's contiguous k order (same as the B fragments):
// lane holds row = lane%16; elements [0..15] = K off..off+15,
// with off = (lane>=16) ? 16 : 0.  rowp points at the start of this lane's row in LDS.
__device__ __forceinline__ v16h load_afrag(const h16* rowp, int off) {
    v8h lo = *(const v8h*)(rowp + off);
    v8h hi = *(const v8h*)(rowp + off + 8);
    return __builtin_shufflevector(lo, hi,
                                   0,1,2,3,4,5,6,7,8,9,10,11,12,13,14,15);
}

__global__ __launch_bounds__(256)
void lstm_fused_kernel(const float* __restrict__ x,     // [B, T, IN]
                       const float* __restrict__ W_ih,  // [4H, IN]
                       const float* __restrict__ W_hh,  // [4H, H]
                       const float* __restrict__ b_ih,  // [4H]
                       const float* __restrict__ b_hh,  // [4H]
                       float* __restrict__ out)         // [B, T, H]
{
    __shared__ __attribute__((aligned(16))) h16   sh_h[MTILE * HDIM];  // resident h (f16)
    __shared__ __attribute__((aligned(16))) h16   sh_x[2][MTILE * 32]; // double-buffered input tile (K padded)
    __shared__ __attribute__((aligned(16))) float sh_g[MTILE * NG];    // raw gates

    const int tid  = threadIdx.x;
    const int w    = tid >> 5;          // wave id 0..7
    const int lane = tid & 31;
    const int lrow = lane & 15;
    const int lhi  = (lane >> 4) & 1;
    const int r0   = blockIdx.x * MTILE;

    // ---- preload time-invariant weight (B-matrix) fragments ----
    // B layout (32x16 f16): lanes 0-15 hold K 0..15, lanes 16-31 hold K 16..31,
    // column = lane%16; element e of v16h -> K = kchunk*32 + lhi*16 + e.
    v16h  bhh[2][2];   // [n-subtile][k-chunk] for W_hh^T (K=64)
    v16h  bxx[2];      // for W_ih^T (K=22 padded to 32)
    float bias[2];
    for (int nl = 0; nl < 2; ++nl) {
        const int col = (2*w + nl) * 16 + lrow;   // gate column 0..255
        for (int kc = 0; kc < 2; ++kc) {
            const int kb = kc*32 + lhi*16;
            v16h tmp;
            #pragma unroll
            for (int e = 0; e < 16; ++e)
                tmp[e] = (h16)W_hh[col * HDIM + kb + e];
            bhh[nl][kc] = tmp;
        }
        {
            const int kb = lhi * 16;
            v16h tmp;
            #pragma unroll
            for (int e = 0; e < 16; ++e) {
                const int k = kb + e;
                tmp[e] = (k < IN_DIM) ? (h16)W_ih[col * IN_DIM + k] : (h16)0.0f;
            }
            bxx[nl] = tmp;
        }
        bias[nl] = b_ih[col] + b_hh[col];
    }

    // ---- per-thread input-tile load slots (branchless; computed once) ----
    // slot i covers element e = tid + 256*i of the padded 16x32 tile.
    size_t xbase[2];   // &x[(r0+row)*T*IN + k], clamped for pad lanes
    bool   xok[2];
    #pragma unroll
    for (int i = 0; i < 2; ++i) {
        const int e = tid + i*256;
        const int row = e >> 5, k = e & 31;
        xok[i]   = (k < IN_DIM);
        xbase[i] = xok[i] ? ((size_t)(r0+row) * TSTEPS * IN_DIM + k) : (size_t)0;
    }

    // elementwise mapping: thread -> (4 rows, 1 h-column)
    const int hcol  = tid & 63;
    const int rbase = (tid >> 6) * 4;
    float cst[4] = {0.f, 0.f, 0.f, 0.f};

    // init h = 0
    #pragma unroll
    for (int j = 0; j < 4; ++j) sh_h[tid*4 + j] = (h16)0.0f;

    // prologue: stage input tile for t = 0 (unconditional loads, data select)
    #pragma unroll
    for (int i = 0; i < 2; ++i) {
        const float v = x[xbase[i]];
        sh_x[0][tid + i*256] = xok[i] ? (h16)v : (h16)0.0f;
    }
    __syncthreads();

    for (int t = 0; t < TSTEPS; ++t) {
        const int cur = t & 1, nxt = cur ^ 1;

        // prefetch next timestep's input tile into registers (unconditional,
        // address clamped at t==T-1; buffer written then is never read).
        const int tn = (t + 1 < TSTEPS) ? (t + 1) : t;
        float pre[2];
        #pragma unroll
        for (int i = 0; i < 2; ++i)
            pre[i] = x[xbase[i] + (size_t)tn * IN_DIM];

        // A fragments (shared h tile + current input tile) from LDS
        const h16* hp = &sh_h[lrow * HDIM];
        const int  ab = lhi * 16;
        const v16h ah0 = load_afrag(hp, ab);        // K 0..31 of h
        const v16h ah1 = load_afrag(hp, 32 + ab);   // K 32..63 of h
        const v16h axt = load_afrag(&sh_x[cur][lrow * 32], ab);

        #pragma unroll
        for (int nl = 0; nl < 2; ++nl) {
            v8f acc = {};
            acc = __builtin_amdgcn_wmma_f32_16x16x32_f16(
                    false, axt, false, bxx[nl],    (short)0, acc, false, false);
            acc = __builtin_amdgcn_wmma_f32_16x16x32_f16(
                    false, ah0, false, bhh[nl][0], (short)0, acc, false, false);
            acc = __builtin_amdgcn_wmma_f32_16x16x32_f16(
                    false, ah1, false, bhh[nl][1], (short)0, acc, false, false);
            // D layout: element r -> row = lhi*8 + r, col = lane%16 (+ tile base)
            const int colg = (2*w + nl)*16 + lrow;
            #pragma unroll
            for (int r = 0; r < 8; ++r) {
                const int row = lhi*8 + r;
                sh_g[row*NG + colg] = acc[r] + bias[nl];
            }
        }
        __syncthreads();

        // elementwise LSTM cell update (gate order: i, f, g, o) — branchless activations
        #pragma unroll
        for (int j = 0; j < 4; ++j) {
            const int row = rbase + j;
            const float gi = sh_g[row*NG +       hcol];
            const float gf = sh_g[row*NG +  64 + hcol];
            const float gg = sh_g[row*NG + 128 + hcol];
            const float go = sh_g[row*NG + 192 + hcol];
            const float iv = fast_sigmoid(gi);
            const float fv = fast_sigmoid(gf);
            const float gv = fast_tanh(gg);
            const float ov = fast_sigmoid(go);
            cst[j] = fv * cst[j] + iv * gv;
            const float hv = ov * fast_tanh(cst[j]);
            out[((size_t)(r0+row)*TSTEPS + t)*HDIM + hcol] = hv;  // coalesced per row
            sh_h[row*HDIM + hcol] = (h16)hv;
        }

        // commit prefetched tile into the other buffer (single loadcnt wait lands here)
        #pragma unroll
        for (int i = 0; i < 2; ++i)
            sh_x[nxt][tid + i*256] = xok[i] ? (h16)pre[i] : (h16)0.0f;
        __syncthreads();
    }
}

extern "C" void kernel_launch(void* const* d_in, const int* in_sizes, int n_in,
                              void* d_out, int out_size, void* d_ws, size_t ws_size,
                              hipStream_t stream) {
    (void)in_sizes; (void)n_in; (void)out_size; (void)d_ws; (void)ws_size;
    const float* x    = (const float*)d_in[0];  // [B, T, IN]
    const float* W_ih = (const float*)d_in[1];  // [4H, IN]
    const float* W_hh = (const float*)d_in[2];  // [4H, H]
    const float* b_ih = (const float*)d_in[3];  // [4H]
    const float* b_hh = (const float*)d_in[4];  // [4H]
    float* out = (float*)d_out;                 // [B, T, H]

    dim3 grid(BATCH / MTILE);   // 256 blocks, each owns 16 batch rows for all T
    dim3 block(256);            // 8 waves (wave32)
    lstm_fused_kernel<<<grid, block, 0, stream>>>(x, W_ih, W_hh, b_ih, b_hh, out);
}
